// PairConnect_73340861546846
// MI455X (gfx1250) — matrix operand order repaired and checked by reading, not compiled, not run on hardware
//
#include <hip/hip_runtime.h>
#include <hip/hip_bf16.h>
#include <math.h>

// CDNA5 / gfx1250: wave32, WMMA 16x16x32 f16->f32, async global->LDS copies,
// permlane cross-lane reductions (no LDS traffic for shuffles).
typedef __attribute__((ext_vector_type(16))) _Float16 v16h;
typedef __attribute__((ext_vector_type(8)))  float    v8f;

#define N_TOK 1024
#define DIM   40
#define BATCH 2
#define TB    512   // 16 waves of 32

__device__ __forceinline__ float gelu_exact(float v) {
    return 0.5f * v * (1.0f + erff(v * 0.70710678118654752f));
}

// ---- cross-lane helpers: VALU permlane instead of ds_bpermute ----
// Sum across the 16 lanes of each half-wave (xor butterfly 1,2,4,8).
__device__ __forceinline__ float row_sum16(float v) {
#if __has_builtin(__builtin_amdgcn_permlane16)
    // selector nibbles give row-relative source lane per lane (lanes0-7 in s1, 8-15 in s2)
    v += __int_as_float(__builtin_amdgcn_permlane16(__float_as_int(v), __float_as_int(v),
                                                    0x67452301u, 0xEFCDAB89u, false, false)); // ^1
    v += __int_as_float(__builtin_amdgcn_permlane16(__float_as_int(v), __float_as_int(v),
                                                    0x54761032u, 0xDCFE98BAu, false, false)); // ^2
    v += __int_as_float(__builtin_amdgcn_permlane16(__float_as_int(v), __float_as_int(v),
                                                    0x32107654u, 0xBA98FEDCu, false, false)); // ^4
    v += __int_as_float(__builtin_amdgcn_permlane16(__float_as_int(v), __float_as_int(v),
                                                    0xFEDCBA98u, 0x76543210u, false, false)); // ^8
#else
    v += __shfl_xor(v, 1, 32);
    v += __shfl_xor(v, 2, 32);
    v += __shfl_xor(v, 4, 32);
    v += __shfl_xor(v, 8, 32);
#endif
    return v;
}
// value from the opposite 16-lane half (lane ^ 16)
__device__ __forceinline__ float half_swap(float v) {
#if __has_builtin(__builtin_amdgcn_permlanex16)
    return __int_as_float(__builtin_amdgcn_permlanex16(__float_as_int(v), __float_as_int(v),
                                                       0x76543210u, 0xFEDCBA98u, false, false));
#else
    return __shfl_xor(v, 16, 32);
#endif
}

__device__ __forceinline__ void wait_async0() {
#if __has_builtin(__builtin_amdgcn_s_wait_asynccnt)
    __builtin_amdgcn_s_wait_asynccnt(0);
#else
    asm volatile("s_wait_asynccnt 0" ::: "memory");
#endif
}

// Async-copy one contiguous 16x40-f32 tile (2560B) into a [16][44] LDS tile.
// 160 lanes each move 16B with GLOBAL_LOAD_ASYNC_TO_LDS_B128 (ASYNCcnt).
// Rows are 176B (16B multiple) so every LDS target is 16B aligned.
__device__ __forceinline__ void async_tile_load(const float* __restrict__ gsrc,
                                                float (*dst)[44], int tid) {
    if (tid < 160) {
        const int r  = tid / 10;
        const int c4 = (tid % 10) * 4;
        const float* src = gsrc + tid * 4;
        unsigned ldsoff = (unsigned)(size_t)&dst[r][c4];  // low 32 bits = LDS offset
        asm volatile("global_load_async_to_lds_b128 %0, %1, off"
                     :: "v"(ldsoff), "v"(src) : "memory");
    }
}

// ---- shared per-j-tile compute: 6x WMMA + online softmax update ----
__device__ __forceinline__ void tile_update(const float (*A1s)[44], const float (*A2s)[44],
                                            const v16h Bf[3][2], const float bt[3],
                                            int w, int jl, int hl, int i, int jt,
                                            float& m, float& Z,
                                            float& acc0, float& acc1, float& acc2) {
    // A fragments: h = gelu(A1[i]+A2[j]) in f16, contiguous k order (same as B):
    // element e -> K = 16*hl+e; second step += 32, K>=40 padded 0.
    v16h a0, a1;
    #pragma unroll
    for (int e = 0; e < 16; ++e) {
        int k = 16 * hl + e;
        a0[e] = (_Float16)gelu_exact(A1s[w][k] + A2s[jl][k]);
        int k2 = k + 32;
        a1[e] = (k2 < DIM) ? (_Float16)gelu_exact(A1s[w][k2] + A2s[jl][k2])
                           : (_Float16)0.0f;
    }

    v8f c0 = {}, c1 = {}, c2 = {};
    c0 = __builtin_amdgcn_wmma_f32_16x16x32_f16(false, a0, false, Bf[0][0], (short)0, c0, false, false);
    c0 = __builtin_amdgcn_wmma_f32_16x16x32_f16(false, a1, false, Bf[0][1], (short)0, c0, false, false);
    c1 = __builtin_amdgcn_wmma_f32_16x16x32_f16(false, a0, false, Bf[1][0], (short)0, c1, false, false);
    c1 = __builtin_amdgcn_wmma_f32_16x16x32_f16(false, a1, false, Bf[1][1], (short)0, c1, false, false);
    c2 = __builtin_amdgcn_wmma_f32_16x16x32_f16(false, a0, false, Bf[2][0], (short)0, c2, false, false);
    c2 = __builtin_amdgcn_wmma_f32_16x16x32_f16(false, a1, false, Bf[2][1], (short)0, c2, false, false);

    // add b2, per-j squared norm reduced over d (16 lanes of the half, VALU permlane)
    float q[8];
    #pragma unroll
    for (int r = 0; r < 8; ++r) {
        float p0 = c0[r] + bt[0], p1 = c1[r] + bt[1], p2 = c2[r] + bt[2];
        c0[r] = p0; c1[r] = p1; c2[r] = p2;
        q[r] = row_sum16(p0 * p0 + p1 * p1 + p2 * p2);
    }

    // online softmax: j = jt*16 + r + 8*hl, mask j > i
    float sv[8]; bool vr[8];
    float tm = -__builtin_inff();
    #pragma unroll
    for (int r = 0; r < 8; ++r) {
        int j = jt * 16 + r + 8 * hl;
        vr[r] = (j <= i);
        sv[r] = sqrtf(q[r]);
        tm = fmaxf(tm, vr[r] ? sv[r] : -__builtin_inff());
    }
    tm = fmaxf(tm, half_swap(tm));              // shared max across halves
    float mnew = fmaxf(m, tm);
    float sc = __expf(m - mnew);                // 0 on the first tile (m=-inf)
    Z *= sc; acc0 *= sc; acc1 *= sc; acc2 *= sc;
    m = mnew;
    #pragma unroll
    for (int r = 0; r < 8; ++r) {
        float wgt = vr[r] ? __expf(sv[r] - m) : 0.0f;
        Z += wgt;
        acc0 = fmaf(wgt, c0[r], acc0);
        acc1 = fmaf(wgt, c1[r], acc1);
        acc2 = fmaf(wgt, c2[r], acc2);
    }
}

__device__ __forceinline__ void load_w2_frags(const float* __restrict__ W2,
                                              const float* __restrict__ b2,
                                              int hl, int jl, v16h Bf[3][2], float bt[3]) {
    // B 32x16 f16, contiguous k order (same as A): element e of v16h -> K = 32*step + 16*hl + e, N = 16*dt + lane%16
    #pragma unroll
    for (int dt = 0; dt < 3; ++dt) {
        #pragma unroll
        for (int s = 0; s < 2; ++s) {
            v16h f;
            #pragma unroll
            for (int e = 0; e < 16; ++e) {
                int K = 32 * s + 16 * hl + e;
                int Nn = 16 * dt + jl;
                f[e] = (K < DIM && Nn < DIM) ? (_Float16)W2[K * DIM + Nn] : (_Float16)0.0f;
            }
            Bf[dt][s] = f;
        }
    }
    #pragma unroll
    for (int dt = 0; dt < 3; ++dt) {
        int d = 16 * dt + jl;
        bt[dt] = (d < DIM) ? b2[d] : 0.0f;
    }
}

__device__ __forceinline__ void write_out(float* __restrict__ out, int b, int i,
                                          int hl, int jl,
                                          float Z, float acc0, float acc1, float acc2) {
    Z    += half_swap(Z);
    acc0 += half_swap(acc0);
    acc1 += half_swap(acc1);
    acc2 += half_swap(acc2);
    if (hl == 0) {
        float inv = 1.0f / Z;
        float* op = out + (size_t)(b * N_TOK + i) * DIM;
        op[jl]      = acc0 * inv;
        op[16 + jl] = acc1 * inv;
        if (jl < 8) op[32 + jl] = acc2 * inv;
    }
}

// ---------- kernel 0: A1 = x@W1top + b1, A2 = x@W1bot (once, into d_ws) ----------
__global__ __launch_bounds__(256)
void precompute_a1a2(const float* __restrict__ x, const float* __restrict__ W1,
                     const float* __restrict__ b1,
                     float* __restrict__ A1g, float* __restrict__ A2g) {
    int idx = blockIdx.x * 256 + threadIdx.x;
    if (idx >= BATCH * N_TOK * DIM) return;
    int n = idx / DIM, d = idx % DIM;
    const float* xr = x + (size_t)n * DIM;
    float s1 = b1[d], s2 = 0.0f;
    #pragma unroll 8
    for (int kk = 0; kk < DIM; ++kk) {
        float xv = xr[kk];
        s1 = fmaf(xv, W1[kk * DIM + d], s1);
        s2 = fmaf(xv, W1[(DIM + kk) * DIM + d], s2);
    }
    A1g[idx] = s1;
    A2g[idx] = s2;
}

// ---------- main fused kernel (precomputed A1/A2 path, async LDS staging) ----------
__global__ __launch_bounds__(TB)
void pairconnect_main(const float* __restrict__ A1g, const float* __restrict__ A2g,
                      const float* __restrict__ W2, const float* __restrict__ b2,
                      float* __restrict__ out) {
    __shared__ __align__(16) float A1s[16][44];
    __shared__ __align__(16) float A2s[16][44];

    const int ib  = blockIdx.x % (N_TOK / 16);
    const int b   = blockIdx.x / (N_TOK / 16);
    const int tid = threadIdx.x;
    const int w   = tid >> 5;
    const int l   = tid & 31;
    const int hl  = l >> 4;
    const int jl  = l & 15;
    const int i   = ib * 16 + w;

    // stage this block's A1 tile (contiguous 2560B) via async copy
    async_tile_load(A1g + (size_t)(b * N_TOK + ib * 16) * DIM, A1s, tid);

    v16h Bf[3][2]; float bt[3];
    load_w2_frags(W2, b2, hl, jl, Bf, bt);
    wait_async0();

    float m = -__builtin_inff(), Z = 0.0f;
    float acc0 = 0.0f, acc1 = 0.0f, acc2 = 0.0f;

    for (int jt = 0; jt <= ib; ++jt) {
        __syncthreads();                // previous A2s fully consumed (also orders A1s)
        async_tile_load(A2g + (size_t)(b * N_TOK + jt * 16) * DIM, A2s, tid);
        wait_async0();
        __syncthreads();

        if (jt + 1 <= ib)               // CDNA5 global_prefetch for next tile
            __builtin_prefetch(A2g + (size_t)(b * N_TOK + (jt + 1) * 16 + jl) * DIM, 0, 0);

        tile_update(A1s, A2s, Bf, bt, w, jl, hl, i, jt, m, Z, acc0, acc1, acc2);
    }
    write_out(out, b, i, hl, jl, Z, acc0, acc1, acc2);
}

// ---------- fallback (no workspace): compute A1/A2 tiles inline ----------
__global__ __launch_bounds__(TB)
void pairconnect_inline(const float* __restrict__ x, const float* __restrict__ W1,
                        const float* __restrict__ b1, const float* __restrict__ W2,
                        const float* __restrict__ b2, float* __restrict__ out) {
    __shared__ float xjs[16][41];
    __shared__ __align__(16) float A1s[16][44];
    __shared__ __align__(16) float A2s[16][44];
    __shared__ float W1bs[40][41];

    const int ib  = blockIdx.x % (N_TOK / 16);
    const int b   = blockIdx.x / (N_TOK / 16);
    const int tid = threadIdx.x;
    const int w   = tid >> 5;
    const int l   = tid & 31;
    const int hl  = l >> 4;
    const int jl  = l & 15;
    const int i   = ib * 16 + w;

    for (int t = tid; t < 40 * 40; t += TB)
        W1bs[t / 40][t % 40] = W1[(40 + t / 40) * 40 + (t % 40)];
    for (int t = tid; t < 16 * 40; t += TB)
        xjs[t / 40][t % 40] = x[(size_t)(b * N_TOK + ib * 16 + t / 40) * 40 + (t % 40)];
    __syncthreads();
    for (int t = tid; t < 16 * 40; t += TB) {
        int ii = t / 40, dd = t % 40;
        float s = b1[dd];
        #pragma unroll 8
        for (int kk = 0; kk < 40; ++kk)
            s += xjs[ii][kk] * W1[kk * 40 + dd];
        A1s[ii][dd] = s;
    }

    v16h Bf[3][2]; float bt[3];
    load_w2_frags(W2, b2, hl, jl, Bf, bt);

    float m = -__builtin_inff(), Z = 0.0f;
    float acc0 = 0.0f, acc1 = 0.0f, acc2 = 0.0f;

    for (int jt = 0; jt <= ib; ++jt) {
        __syncthreads();
        for (int t = tid; t < 16 * 40; t += TB)
            xjs[t / 40][t % 40] = x[(size_t)(b * N_TOK + jt * 16 + t / 40) * 40 + (t % 40)];
        __syncthreads();
        for (int t = tid; t < 16 * 40; t += TB) {
            int jj = t / 40, dd = t % 40;
            float s = 0.0f;
            #pragma unroll 8
            for (int kk = 0; kk < 40; ++kk)
                s += xjs[jj][kk] * W1bs[kk][dd];
            A2s[jj][dd] = s;
        }
        __syncthreads();

        if (jt + 1 <= ib)
            __builtin_prefetch(&x[(size_t)(b * N_TOK + (jt + 1) * 16 + jl) * 40], 0, 0);

        tile_update(A1s, A2s, Bf, bt, w, jl, hl, i, jt, m, Z, acc0, acc1, acc2);
    }
    write_out(out, b, i, hl, jl, Z, acc0, acc1, acc2);
}

extern "C" void kernel_launch(void* const* d_in, const int* in_sizes, int n_in,
                              void* d_out, int out_size, void* d_ws, size_t ws_size,
                              hipStream_t stream) {
    (void)in_sizes; (void)n_in; (void)out_size;
    const float* x  = (const float*)d_in[0];
    const float* W1 = (const float*)d_in[1];
    const float* b1 = (const float*)d_in[2];
    const float* W2 = (const float*)d_in[3];
    const float* b2 = (const float*)d_in[4];
    float* out = (float*)d_out;

    const size_t elems = (size_t)BATCH * N_TOK * DIM;     // 81920
    const size_t need  = 2 * elems * sizeof(float);       // 640 KB
    dim3 grid(BATCH * (N_TOK / 16));                      // 128 blocks

    if (d_ws != nullptr && ws_size >= need) {
        float* A1g = (float*)d_ws;
        float* A2g = A1g + elems;
        precompute_a1a2<<<(int)((elems + 255) / 256), 256, 0, stream>>>(x, W1, b1, A1g, A2g);
        pairconnect_main<<<grid, TB, 0, stream>>>(A1g, A2g, W2, b2, out);
    } else {
        pairconnect_inline<<<grid, TB, 0, stream>>>(x, W1, b1, W2, b2, out);
    }
}
